// AbsoluteMinimalModel_12068858102143
// MI455X (gfx1250) — compile-verified, matrix operand order checked by reading, not run on hardware
//
#include <hip/hip_runtime.h>
#include <hip/hip_bf16.h>

typedef __attribute__((ext_vector_type(16))) _Float16 v16h;
typedef __attribute__((ext_vector_type(8)))  _Float16 v8h;
typedef __attribute__((ext_vector_type(8)))  float    v8f;

#define NTOK   4096        // B*S = 2*2048
#define DMODEL 1024
#define V1N    200
#define V2N    160
#define NLAYER 8
#define EPSR   1e-6f
#define MT_PER_WAVE 8      // M-tiles per wave in stage-2 GEMM

// ---------------------------------------------------------------------------
// helpers
// ---------------------------------------------------------------------------
__device__ __forceinline__ float wave_reduce_add(float v) {
    #pragma unroll
    for (int off = 16; off > 0; off >>= 1) v += __shfl_xor(v, off, 32);
    return v;
}

// Load a 16x32 f16 tile (row-major, row stride 32 f16) into WMMA A/B fragment.
// Layout (CDNA5 16-bit A 16x32): lane L<16 -> row L, K = 0..7 (v0-3) & 16..23 (v4-7)
//                                lane L>=16 -> row L-16, K = 8..15 & 24..31
__device__ __forceinline__ v16h load_frag16x32(const _Float16* __restrict__ base) {
    int lane = threadIdx.x & 31;
    int r    = lane & 15;
    int kh   = (lane >> 4) & 1;          // 0 -> K base 0, 1 -> K base 8
    const _Float16* p = base + r * 32 + kh * 8;
    v8h lo = *(const v8h*)(p);           // K = kh*8 .. kh*8+7       -> VGPR 0-3
    v8h hi = *(const v8h*)(p + 16);      // K = 16+kh*8 .. 16+kh*8+7 -> VGPR 4-7
    v16h f;
    #pragma unroll
    for (int i = 0; i < 8; i++) { f[i] = lo[i]; f[8 + i] = hi[i]; }
    return f;
}

__device__ __forceinline__ float gelu_tanh(float h) {
    float c = 0.7978845608028654f * (h + 0.044715f * h * h * h);
    return 0.5f * h * (1.0f + tanhf(c));
}

// ---------------------------------------------------------------------------
// K0: core2 -> f16 (B matrix for stage-2 WMMA)
// ---------------------------------------------------------------------------
__global__ void k_cvt_core2(const float* __restrict__ core2, _Float16* __restrict__ c2h) {
    int i = blockIdx.x * blockDim.x + threadIdx.x;
    if (i < V2N * 32) c2h[i] = (_Float16)core2[i];
}

// ---------------------------------------------------------------------------
// K1: TT-rank-1 embedding. x[t, d1*32+d2] = core1[i1,d1]*core2[i2,d2]
// ---------------------------------------------------------------------------
__global__ void k_embed(const int* __restrict__ ids, const float* __restrict__ core1,
                        const float* __restrict__ core2, float* __restrict__ x) {
    int t  = blockIdx.x;
    int id = ids[t];
    int i1 = id / V2N, i2 = id % V2N;
    int d0 = threadIdx.x * 4;
    float* xp = x + (size_t)t * DMODEL;
    #pragma unroll
    for (int j = 0; j < 4; j++) {
        int d = d0 + j;
        xp[d] = core1[i1 * 32 + (d >> 5)] * core2[i2 * 32 + (d & 31)];
    }
}

// ---------------------------------------------------------------------------
// K2: rmsnorm (x -> xn) ; one token per block, 256 threads x 4 elems
// ---------------------------------------------------------------------------
__global__ void k_rmsnorm(const float* __restrict__ x, const float* __restrict__ w,
                          float* __restrict__ xn) {
    __shared__ float sm[8];
    int t = blockIdx.x, tid = threadIdx.x;
    const float4* xp = (const float4*)(x + (size_t)t * DMODEL);
    float4 v = xp[tid];
    float ss = v.x * v.x + v.y * v.y + v.z * v.z + v.w * v.w;
    ss = wave_reduce_add(ss);
    if ((tid & 31) == 0) sm[tid >> 5] = ss;
    __syncthreads();
    float tot = 0.f;
    #pragma unroll
    for (int i = 0; i < 8; i++) tot += sm[i];
    float scale = rsqrtf(tot / (float)DMODEL + EPSR);
    float4 wv = ((const float4*)w)[tid];
    float4 o;
    o.x = v.x * scale * wv.x; o.y = v.y * scale * wv.y;
    o.z = v.z * scale * wv.z; o.w = v.w * scale * wv.w;
    ((float4*)(xn + (size_t)t * DMODEL))[tid] = o;
}

// ---------------------------------------------------------------------------
// K3: rank-1 semiseparable scan, chunked (8 chunks of 256) for parallelism.
// x += u * scan(a = sigmoid(lam), v * xn).  Block: 32 channels x 8 chunks.
// ---------------------------------------------------------------------------
__global__ void k_scan(const float* __restrict__ xn, const float* __restrict__ lam,
                       const float* __restrict__ u, const float* __restrict__ v,
                       float* __restrict__ x) {
    __shared__ float hend[256];
    __shared__ float hstart[256];
    int tid   = threadIdx.x;
    int cl    = tid & 31;           // channel within block
    int chunk = tid >> 5;           // 0..7
    int c = blockIdx.x * 32 + cl;   // 0..2047 : channel = (b, d)
    int b = c >> 10, d = c & 1023;
    float a  = 1.f / (1.f + __expf(-lam[d]));
    float vi = v[d], uo = u[d];
    size_t base = (size_t)b * 2048 * DMODEL + d;
    int s0 = chunk * 256;
    // phase 1: local scan with h=0, record carry
    float h = 0.f;
    for (int s = s0; s < s0 + 256; s++)
        h = a * h + vi * xn[base + (size_t)s * DMODEL];
    hend[chunk * 32 + cl] = h;
    __syncthreads();
    // phase 2: serial carry combine over 8 chunks (threads with chunk==0)
    if (tid < 32) {
        float ap = a;                                    // a^256 via 8 squarings
        #pragma unroll
        for (int i = 0; i < 8; i++) ap = ap * ap;
        float H = 0.f;
        #pragma unroll
        for (int j = 0; j < 8; j++) {
            hstart[j * 32 + cl] = H;
            H = ap * H + hend[j * 32 + cl];
        }
    }
    __syncthreads();
    // phase 3: rescan with correct h_start, emit outputs
    h = hstart[chunk * 32 + cl];
    for (int s = s0; s < s0 + 256; s++) {
        h = a * h + vi * xn[base + (size_t)s * DMODEL];
        x[base + (size_t)s * DMODEL] += uo * h;
    }
}

// ---------------------------------------------------------------------------
// K4: fused rmsnorm + rank-2 FFN residual. One token per block.
// r_j = rsqrt(ms) * sum_d x_d*nw_d*w1[d,j] ; x += gelu(r0)*w2[0,:] + gelu(r1)*w2[1,:]
// ---------------------------------------------------------------------------
__global__ void k_ffn(float* __restrict__ x, const float* __restrict__ n2w,
                      const float* __restrict__ w1, const float* __restrict__ w2) {
    __shared__ float sm[24];
    int t = blockIdx.x, tid = threadIdx.x;
    float4 v  = ((const float4*)(x + (size_t)t * DMODEL))[tid];
    float4 wv = ((const float4*)n2w)[tid];
    float4 a0 = ((const float4*)(w1 + tid * 8))[0];   // w1[d,0..1] interleaved
    float4 a1 = ((const float4*)(w1 + tid * 8))[1];
    float hx = v.x * wv.x, hy = v.y * wv.y, hz = v.z * wv.z, hw = v.w * wv.w;
    float ss = v.x * v.x + v.y * v.y + v.z * v.z + v.w * v.w;
    float s0 = hx * a0.x + hy * a0.z + hz * a1.x + hw * a1.z;
    float s1 = hx * a0.y + hy * a0.w + hz * a1.y + hw * a1.w;
    ss = wave_reduce_add(ss); s0 = wave_reduce_add(s0); s1 = wave_reduce_add(s1);
    if ((tid & 31) == 0) {
        int w = tid >> 5;
        sm[w] = ss; sm[8 + w] = s0; sm[16 + w] = s1;
    }
    __syncthreads();
    float tss = 0.f, ts0 = 0.f, ts1 = 0.f;
    #pragma unroll
    for (int i = 0; i < 8; i++) { tss += sm[i]; ts0 += sm[8 + i]; ts1 += sm[16 + i]; }
    float scale = rsqrtf(tss / (float)DMODEL + EPSR);
    float g0 = gelu_tanh(scale * ts0);
    float g1 = gelu_tanh(scale * ts1);
    float4 b0 = ((const float4*)(w2 + 0))[tid];        // w2[0, d]
    float4 b1 = ((const float4*)(w2 + DMODEL))[tid];   // w2[1, d]
    v.x += g0 * b0.x + g1 * b1.x;  v.y += g0 * b0.y + g1 * b1.y;
    v.z += g0 * b0.z + g1 * b1.z;  v.w += g0 * b0.w + g1 * b1.w;
    ((float4*)(x + (size_t)t * DMODEL))[tid] = v;
}

// ---------------------------------------------------------------------------
// K5: final rmsnorm -> f16
// ---------------------------------------------------------------------------
__global__ void k_finalnorm(const float* __restrict__ x, const float* __restrict__ w,
                            _Float16* __restrict__ xf) {
    __shared__ float sm[8];
    int t = blockIdx.x, tid = threadIdx.x;
    float4 v = ((const float4*)(x + (size_t)t * DMODEL))[tid];
    float ss = v.x * v.x + v.y * v.y + v.z * v.z + v.w * v.w;
    ss = wave_reduce_add(ss);
    if ((tid & 31) == 0) sm[tid >> 5] = ss;
    __syncthreads();
    float tot = 0.f;
    #pragma unroll
    for (int i = 0; i < 8; i++) tot += sm[i];
    float scale = rsqrtf(tot / (float)DMODEL + EPSR);
    float4 wv = ((const float4*)w)[tid];
    _Float16* o = xf + (size_t)t * DMODEL + tid * 4;
    o[0] = (_Float16)(v.x * scale * wv.x);
    o[1] = (_Float16)(v.y * scale * wv.y);
    o[2] = (_Float16)(v.z * scale * wv.z);
    o[3] = (_Float16)(v.w * scale * wv.w);
}

// ---------------------------------------------------------------------------
// K6: stage 1 of TT projection: S[t,v1,d2] = sum_d1 core1[v1,d1]*xf[t,d1*32+d2]
// One thread per (t, d2); loops all 200 v1 rows, so xf16 is read exactly once
// overall and core1 accesses are wave-uniform (scalar-load path).
// ---------------------------------------------------------------------------
__global__ void k_stage1(const _Float16* __restrict__ xf, const float* __restrict__ core1,
                         _Float16* __restrict__ S) {
    int gid = blockIdx.x * blockDim.x + threadIdx.x;   // 0 .. 131071
    int d2  = gid & 31;
    int t   = gid >> 5;
    const _Float16* xp = xf + (size_t)t * DMODEL + d2;
    float xv[32];
    #pragma unroll
    for (int d1 = 0; d1 < 32; d1++) xv[d1] = (float)xp[d1 * 32];
    _Float16* sp = S + (size_t)t * (V1N * 32) + d2;
    #pragma unroll 4
    for (int v1 = 0; v1 < V1N; v1++) {
        const float* cp = core1 + v1 * 32;
        float acc = 0.f;
        #pragma unroll
        for (int d1 = 0; d1 < 32; d1++) acc += xv[d1] * cp[d1];
        sp[(size_t)v1 * 32] = (_Float16)acc;
    }
}

// ---------------------------------------------------------------------------
// K7: stage 2 WMMA GEMM: [819200 x 32] @ [32 x 160] -> logits [819200 x 160].
// K=32 == one v_wmma_f32_16x16x32_f16 per 16x16 tile. Each wave keeps the 10
// core2 B-fragments resident and loops over 8 M-tiles (amortizes B traffic
// 8x -> ~64 MB L2 reads vs 512 MB), double-buffering the next A-fragment so
// its load overlaps this tile's WMMAs + non-temporal streaming stores.
// ---------------------------------------------------------------------------
__global__ void k_stage2_wmma(const _Float16* __restrict__ S,
                              const _Float16* __restrict__ c2h,
                              float* __restrict__ out) {
    int wid  = (blockIdx.x * blockDim.x + threadIdx.x) >> 5;   // 0 .. 6399
    int lane = threadIdx.x & 31;
    int r = lane & 15;            // N column within tile
    int g = (lane >> 4) & 1;      // 0: rows i, 1: rows i+8
    v16h b[10];
    #pragma unroll
    for (int nt = 0; nt < 10; nt++) b[nt] = load_frag16x32(c2h + nt * 512);
    size_t mt0 = (size_t)wid * MT_PER_WAVE;
    v16h a = load_frag16x32(S + mt0 * 512);
    for (int m = 0; m < MT_PER_WAVE; m++) {
        // prefetch next A tile (last iteration touches the c2h region that
        // directly follows S in the workspace; value is discarded)
        v16h an = load_frag16x32(S + (mt0 + m + 1) * 512);
        size_t row0 = (mt0 + m) * 16;
        #pragma unroll
        for (int nt = 0; nt < 10; nt++) {
            v8f c = {};
            c = __builtin_amdgcn_wmma_f32_16x16x32_f16(
                    false, a, false, b[nt], (short)0, c, false, false);
            float* o = out + row0 * V2N + nt * 16 + r;
            #pragma unroll
            for (int i = 0; i < 8; i++) {
                __builtin_nontemporal_store(c[i], o + (size_t)(i + g * 8) * V2N);
            }
        }
        a = an;
    }
}

// ---------------------------------------------------------------------------
// launch
// ---------------------------------------------------------------------------
extern "C" void kernel_launch(void* const* d_in, const int* in_sizes, int n_in,
                              void* d_out, int out_size, void* d_ws, size_t ws_size,
                              hipStream_t stream) {
    const int*   ids    = (const int*)  d_in[0];
    const float* core1  = (const float*)d_in[1];   // [200,32]
    const float* core2  = (const float*)d_in[2];   // [160,32]
    const float* lam    = (const float*)d_in[3];   // [8,1024]
    const float* u      = (const float*)d_in[4];
    const float* v      = (const float*)d_in[5];
    const float* w1     = (const float*)d_in[6];   // [8,1024,2]
    const float* w2     = (const float*)d_in[7];   // [8,2,1024]
    const float* n1w    = (const float*)d_in[8];
    const float* n2w    = (const float*)d_in[9];
    const float* fnw    = (const float*)d_in[10];
    float* out = (float*)d_out;

    char* ws = (char*)d_ws;
    float*     x    = (float*)    (ws + 0);            // 16.78 MB
    float*     xn   = (float*)    (ws + 16777216);     // 16.78 MB
    _Float16*  xf16 = (_Float16*) (ws + 33554432);     //  8.39 MB
    _Float16*  S    = (_Float16*) (ws + 41943040);     // 52.43 MB
    _Float16*  c2h  = (_Float16*) (ws + 94371840);     // 10 KB

    k_cvt_core2<<<20, 256, 0, stream>>>(core2, c2h);
    k_embed<<<NTOK, 256, 0, stream>>>(ids, core1, core2, x);

    for (int l = 0; l < NLAYER; l++) {
        k_rmsnorm<<<NTOK, 256, 0, stream>>>(x, n1w + l * DMODEL, xn);
        k_scan<<<64, 256, 0, stream>>>(xn, lam + l * DMODEL, u + l * DMODEL,
                                       v + l * DMODEL, x);
        k_ffn<<<NTOK, 256, 0, stream>>>(x, n2w + l * DMODEL,
                                        w1 + l * DMODEL * 2, w2 + l * 2 * DMODEL);
    }

    k_finalnorm<<<NTOK, 256, 0, stream>>>(x, fnw, xf16);
    // 4096 tokens * 32 d2 = 131072 threads
    k_stage1<<<512, 256, 0, stream>>>(xf16, core1, S);
    // 819200/16 = 51200 M-tiles, 8 per wave -> 6400 waves -> 800 blocks
    k_stage2_wmma<<<800, 256, 0, stream>>>(S, c2h, out);
}
